// ClientNet_53343493817046
// MI455X (gfx1250) — matrix operand order repaired and checked by reading, not compiled, not run on hardware
//
#include <hip/hip_runtime.h>
#include <hip/hip_bf16.h>

typedef __attribute__((ext_vector_type(16))) _Float16 v16h;
typedef __attribute__((ext_vector_type(8)))  _Float16 v8h;
typedef __attribute__((ext_vector_type(8)))  float    v8f;

#define WMMA16(A,B,C) __builtin_amdgcn_wmma_f32_16x16x32_f16(false,(A),false,(B),(short)0,(C),false,false)

static __device__ __forceinline__ v16h cat8(v8h lo, v8h hi) {
    return __builtin_shufflevector(lo, hi, 0,1,2,3,4,5,6,7,8,9,10,11,12,13,14,15);
}

// gfx1250 async global->LDS copy (16B per lane). lds_byte = LDS byte offset
// (low 32 bits of a generic pointer to shared memory, per the aperture rules).
static __device__ __forceinline__ void async_ld16(unsigned lds_byte, unsigned long long gaddr) {
    asm volatile("global_load_async_to_lds_b128 %0, %1, off"
                 :: "v"(lds_byte), "v"(gaddr) : "memory");
}
static __device__ __forceinline__ void async_wait0() {
#if __has_builtin(__builtin_amdgcn_s_wait_asynccnt)
    __builtin_amdgcn_s_wait_asynccnt(0);
#else
    asm volatile("s_wait_asynccnt 0" ::: "memory");
#endif
}
static __device__ __forceinline__ unsigned lds_off(const void* p) {
    return (unsigned)(size_t)p;     // generic addr low 32 bits == LDS byte address
}

// ---------------- workspace layout (bytes) ----------------
static constexpr size_t O_W1S  = 0;                           // 1024 f16  (conv1 B-frags: 2 ntiles x 512)
static constexpr size_t O_B1   = 4096;                        // 32 f32
static constexpr size_t O_B2S  = 8192;                        // 38912 f16 (conv2 B-frags: 19 ksteps x 4 ntiles x 512)
static constexpr size_t O_B2   = O_B2S + 77824;               // 64 f32   (=86016)
static constexpr size_t O_BF1S = 90112;                       // 409600 f16 (fc1: 25 ksteps x 32 ntiles x 512)
static constexpr size_t O_BF1  = O_BF1S + 819200;             // 512 f32  (=909312)
static constexpr size_t O_BF2S = 913408;                      // 8192 f16 (fc2: 16 ksteps x 1 ntile x 512)
static constexpr size_t O_BF2  = O_BF2S + 16384;              // 16 f32   (=929792)
static constexpr size_t O_Y1   = 933888;                      // 8192*3456 f16  [b][12][12][24] (c 20..23 zero)
static constexpr size_t O_Y2   = O_Y1 + (size_t)8192*3456*2;  // 8192*800  f16  [b][800] (torch flatten order)

// =========================================================
// Kernel 1: ps-average all weights, emit WMMA-B-frag swizzled f16 + f32 biases
// B-frag layout per 32x16 (KxN) block bb: elem(bb,l,j) = B[kbase+(l/16)*8+(j%8)+(j/8)*16][ntile*16+l%16]
// =========================================================
__global__ __launch_bounds__(256) void prep_kernel(
    const float* __restrict__ ps,
    const float* __restrict__ c1w, const float* __restrict__ c1b,
    const float* __restrict__ c2w, const float* __restrict__ c2b,
    const float* __restrict__ f1w, const float* __restrict__ f1b,
    const float* __restrict__ f2w, const float* __restrict__ f2b,
    _Float16* __restrict__ w1s, float* __restrict__ b1,
    _Float16* __restrict__ b2s, float* __restrict__ b2,
    _Float16* __restrict__ bf1s, float* __restrict__ bf1,
    _Float16* __restrict__ bf2s, float* __restrict__ bf2)
{
    __shared__ float psl[16];
    if (threadIdx.x < 16) psl[threadIdx.x] = ps[threadIdx.x];
    __syncthreads();
    long e = (long)blockIdx.x * 256 + threadIdx.x;

    if (e < 1024) {                                  // conv1 weights, Nt=2, 1 kstep, k=ky*5+kx
        int bb = (int)(e >> 9), l = (int)((e >> 4) & 31), j = (int)(e & 15);
        int k = ((l >> 4) << 3) + (j & 7) + ((j >> 3) << 4);
        int n = (bb << 4) + (l & 15);
        float s = 0.f;
        if (k < 25 && n < 20)
            for (int i = 0; i < 16; ++i) s += psl[i] * c1w[(i*20 + n)*25 + k];
        w1s[e] = (_Float16)s; return;
    }
    e -= 1024;
    if (e < 32) { float s=0.f; if (e<20) for(int i=0;i<16;++i) s+=psl[i]*c1b[i*20+e]; b1[e]=s; return; }
    e -= 32;
    if (e < 38912) {                                 // conv2 weights, Nt=4, 19 ksteps, k=tap*24+c (c padded to 24)
        int bb=(int)(e>>9), l=(int)((e>>4)&31), j=(int)(e&15);
        int kstep=bb>>2, ntile=bb&3;
        int k=kstep*32 + ((l>>4)<<3) + (j&7) + ((j>>3)<<4);
        int n=ntile*16 + (l&15);
        int tap=k/24, c=k-tap*24;
        float s=0.f;
        if (tap < 25 && c < 20 && n < 50)
            for(int i=0;i<16;++i) s += psl[i]*c2w[((i*50+n)*20+c)*25 + tap];
        b2s[e]=(_Float16)s; return;
    }
    e -= 38912;
    if (e < 64) { float s=0.f; if (e<50) for(int i=0;i<16;++i) s+=psl[i]*c2b[i*50+e]; b2[e]=s; return; }
    e -= 64;
    if (e < 409600) {                                // fc1 weights, Nt=32, 25 ksteps
        int bb=(int)(e>>9), l=(int)((e>>4)&31), j=(int)(e&15);
        int kstep=bb>>5, ntile=bb&31;
        int k=kstep*32 + ((l>>4)<<3) + (j&7) + ((j>>3)<<4);
        int n=ntile*16 + (l&15);
        float s=0.f;
        if (n < 500)
            for(int i=0;i<16;++i) s += psl[i]*f1w[(long)(i*500+n)*800 + k];
        bf1s[e]=(_Float16)s; return;
    }
    e -= 409600;
    if (e < 512){ float s=0.f; if(e<500) for(int i=0;i<16;++i) s+=psl[i]*f1b[i*500+e]; bf1[e]=s; return; }
    e -= 512;
    if (e < 8192){                                   // fc2 weights, Nt=1, 16 ksteps
        int bb=(int)(e>>9), l=(int)((e>>4)&31), j=(int)(e&15);
        int k=bb*32 + ((l>>4)<<3) + (j&7) + ((j>>3)<<4);
        int n=l&15;
        float s=0.f;
        if (k < 500 && n < 10)
            for(int i=0;i<16;++i) s += psl[i]*f2w[(i*10+n)*500 + k];
        bf2s[e]=(_Float16)s; return;
    }
    e -= 8192;
    if (e < 16){ float s=0.f; if(e<10) for(int i=0;i<16;++i) s+=psl[i]*f2b[i*10+e]; bf2[e]=s; return; }
}
static constexpr int PREP_ELEMS = 1024+32+38912+64+409600+512+8192+16; // 458352

// =========================================================
// Kernel 2: conv1(5x5, cin=1) + bias + relu + 2x2 maxpool. One block per image.
// Tap offsets hoisted out of the tile loop; invalid k -> guaranteed-zero LDS slot.
// Output y1 [b][12][12][24] f16, channels 20..23 zero.
// =========================================================
__global__ __launch_bounds__(256) void conv1_kernel(
    const float* __restrict__ x, const _Float16* __restrict__ w1s,
    const float* __restrict__ b1, _Float16* __restrict__ y1)
{
    __shared__ _Float16 xh[1600];                 // 784 image + zero tail
    __shared__ __align__(32) _Float16 wl[1024];
    __shared__ _Float16 conv[576*20];
    __shared__ float b1l[20];

    int b = blockIdx.x, t = threadIdx.x;
    for (int i = t; i < 1600; i += 256)
        xh[i] = (i < 784) ? (_Float16)x[(size_t)b*784 + i] : (_Float16)0.f;
    { // weights: async global->LDS (2 KB = 128 x 16B)
        unsigned lw = lds_off(wl);
        if (t < 128)
            async_ld16(lw + t*16u, (unsigned long long)(size_t)w1s + (unsigned long long)t*16u);
    }
    if (t < 20) b1l[t] = b1[t];
    async_wait0();
    __syncthreads();

    int wave = t >> 5, lane = t & 31;
    int row = lane & 15, half = lane >> 4, koff = half * 8, n16 = lane & 15;

    v16h bfr0 = *(const v16h*)(wl + lane*16);
    v16h bfr1 = *(const v16h*)(wl + 512 + lane*16);

    // per-lane tap offsets, computed once (ZOFF=790: base+790 is in the zero tail for any base<=779)
    int offs[16];
#pragma unroll
    for (int i = 0; i < 8; ++i) {
        int k = koff + i;
        offs[i]   = (k  < 25) ? (k /5)*28 + (k %5) : 790;
        int k2 = koff + 16 + i;
        offs[8+i] = (k2 < 25) ? (k2/5)*28 + (k2%5) : 790;
    }

    for (int tile = wave; tile < 36; tile += 8) {
        int p = tile*16 + row, oy = p/24, ox = p%24;
        int base = oy*28 + ox;
        v16h a;
#pragma unroll
        for (int i = 0; i < 16; ++i) a[i] = xh[base + offs[i]];
#pragma unroll
        for (int g = 0; g < 2; ++g) {
            v8f acc = {};
            acc = WMMA16(a, (g ? bfr1 : bfr0), acc);
            int ch = g*16 + n16;
            if (ch < 20) {
                int pr = tile*16 + half*8;
#pragma unroll
                for (int r = 0; r < 8; ++r) {
                    float v = acc[r] + b1l[ch];
                    conv[(pr + r)*20 + ch] = (_Float16)(v > 0.f ? v : 0.f);
                }
            }
        }
    }
    __syncthreads();
    // 2x2 maxpool 24x24 -> 12x12, write stride-24 channel-last (c>=20 zero)
    for (int q = t; q < 3456; q += 256) {
        int c = q % 24, px = (q/24) % 12, py = q / 288;
        _Float16 outv = (_Float16)0.f;
        if (c < 20) {
            float m = -1e30f;
#pragma unroll
            for (int dy = 0; dy < 2; ++dy)
#pragma unroll
                for (int dx = 0; dx < 2; ++dx) {
                    float v = (float)conv[((2*py+dy)*24 + (2*px+dx))*20 + c];
                    m = v > m ? v : m;
                }
            outv = (_Float16)m;
        }
        y1[(size_t)b*3456 + q] = outv;
    }
}

// =========================================================
// Kernel 3: conv2(5x5, 20->50) + bias + relu + pool. TWO images per block
// (weight staging amortized, async global->LDS). 4 waves per image.
// K = 25 taps * 24 ch = 600 pad 608 (19 ksteps). A-frag = two aligned v8h LDS loads.
// =========================================================
__global__ __launch_bounds__(256) void conv2_kernel(
    const _Float16* __restrict__ y1, const _Float16* __restrict__ b2s,
    const float* __restrict__ b2, _Float16* __restrict__ y2)
{
    __shared__ __align__(32) _Float16 a1p[2][3840];   // [img][pixel*24], rows 144..159 zero
    __shared__ __align__(32) _Float16 wl[38912];      // swizzled weights, 76 KB
    __shared__ _Float16 conv[2][3200];                // [img][64 px][50 ch]
    __shared__ float b2l[64];

    int b0 = blockIdx.x*2, t = threadIdx.x;
    { // async-stage weights + both images; zero pad rows with plain LDS stores
        unsigned lw = lds_off(wl);
        unsigned long long gw = (unsigned long long)(size_t)b2s;
        for (int i = t; i < 4864; i += 256)
            async_ld16(lw + (unsigned)i*16u, gw + (unsigned long long)i*16u);
        unsigned la = lds_off(a1p);
        for (int i = t; i < 864; i += 256) {
            int img = i / 432, ii = i - img*432;
            async_ld16(la + (unsigned)(img*7680 + ii*16),
                       (unsigned long long)(size_t)(y1 + (size_t)(b0+img)*3456) + (unsigned long long)ii*16u);
        }
        uint4 z; z.x = z.y = z.z = z.w = 0u;
        if (t < 96) {                                 // 2 imgs x 48 x 16B of pad rows
            int img = t / 48, ii = t - img*48;
            ((uint4*)a1p[img])[432 + ii] = z;
        }
        if (t < 64) b2l[t] = b2[t];
    }
    async_wait0();
    __syncthreads();

    int wave = t >> 5, lane = t & 31;
    int row = lane & 15, half = lane >> 4, koff = half * 8, n16 = lane & 15;
    int img = wave >> 2, mtile = wave & 3;
    int p = mtile*16 + row, oy = p >> 3, ox = p & 7;
    int pixbase = oy*12 + ox;
    const _Float16* ai = a1p[img];

    v8f acc[4] = {{}, {}, {}, {}};
    for (int kstep = 0; kstep < 19; ++kstep) {
        int k0 = kstep*32 + koff;              // multiple of 8 -> 16B-aligned runs
        int tap0 = k0/24, c0 = k0 - tap0*24;
        int k1 = k0 + 16;
        int tap1 = k1/24, c1 = k1 - tap1*24;
        int a0 = (pixbase + (tap0/5)*12 + (tap0%5))*24 + c0;
        int a1 = (pixbase + (tap1/5)*12 + (tap1%5))*24 + c1;
        v16h a = cat8(*(const v8h*)(ai + a0), *(const v8h*)(ai + a1));
#pragma unroll
        for (int u = 0; u < 4; ++u) {
            v16h bb = *(const v16h*)(wl + (kstep*4 + u)*512 + lane*16);
            acc[u] = WMMA16(a, bb, acc[u]);
        }
    }
#pragma unroll
    for (int u = 0; u < 4; ++u) {
        int ch = u*16 + n16;
        if (ch < 50) {
            int pr = mtile*16 + half*8;
#pragma unroll
            for (int r = 0; r < 8; ++r) {
                float v = acc[u][r] + b2l[ch];
                conv[img][(pr + r)*50 + ch] = (_Float16)(v > 0.f ? v : 0.f);
            }
        }
    }
    __syncthreads();
    // pool 8x8 -> 4x4; emit torch-flatten order f = c*16 + py*4 + px
    for (int i = t; i < 1600; i += 256) {
        int im = i / 800, q = i - im*800;
        int c = q >> 4, rem = q & 15, py = rem >> 2, px = rem & 3;
        float m = -1e30f;
#pragma unroll
        for (int dy = 0; dy < 2; ++dy)
#pragma unroll
            for (int dx = 0; dx < 2; ++dx) {
                float v = (float)conv[im][((2*py+dy)*8 + (2*px+dx))*50 + c];
                m = v > m ? v : m;
            }
        y2[(size_t)(b0+im)*800 + q] = (_Float16)m;
    }
}

// =========================================================
// Kernel 4: fc1(800->500)+relu then fc2(500->10)+bias, fused. 32 batch rows per block
// (2 mtiles -> each B-fragment register feeds two WMMAs). fc2 on waves 0-1.
// =========================================================
__global__ __launch_bounds__(256) void fc_kernel(
    const _Float16* __restrict__ y2, const _Float16* __restrict__ bf1s,
    const float* __restrict__ bf1, const _Float16* __restrict__ bf2s,
    const float* __restrict__ bf2, float* __restrict__ out)
{
    __shared__ __align__(16) _Float16 at[32*808];   // A tile, padded stride
    __shared__ __align__(16) _Float16 h1[32*520];   // fc1 activations (f16), padded stride
    __shared__ float bf1l[512];
    __shared__ float bf2l[16];

    int t = threadIdx.x, b0 = blockIdx.x * 32;
    { // async-stage A tile (32 rows x 100 x 16B, padded LDS row stride 1616B)
        unsigned la = lds_off(at);
        for (int i = t; i < 3200; i += 256) {
            int r = i / 100, c = i - r*100;
            async_ld16(la + (unsigned)(r*1616 + c*16),
                       (unsigned long long)(size_t)(y2 + (size_t)(b0 + r)*800) + (unsigned long long)c*16u);
        }
        for (int i = t; i < 512; i += 256) bf1l[i] = bf1[i];
        if (t < 16) bf2l[t] = bf2[t];
    }
    async_wait0();
    __syncthreads();

    int wave = t >> 5, lane = t & 31;
    int row = lane & 15, half = lane >> 4, koff = half * 8, n16 = lane & 15;

    v8f acc[2][4] = {{{}, {}, {}, {}}, {{}, {}, {}, {}}};
    for (int kstep = 0; kstep < 25; ++kstep) {
        const _Float16* ap0 = at + row*808 + kstep*32 + koff;
        const _Float16* ap1 = ap0 + 16*808;
        v16h a0 = cat8(*(const v8h*)ap0, *(const v8h*)(ap0 + 16));
        v16h a1 = cat8(*(const v8h*)ap1, *(const v8h*)(ap1 + 16));
#pragma unroll
        for (int u = 0; u < 4; ++u) {
            int ntile = wave*4 + u;
            v16h bb = *(const v16h*)(bf1s + (size_t)(kstep*32 + ntile)*512 + lane*16);
            acc[0][u] = WMMA16(a0, bb, acc[0][u]);
            acc[1][u] = WMMA16(a1, bb, acc[1][u]);
        }
    }
#pragma unroll
    for (int m = 0; m < 2; ++m)
#pragma unroll
        for (int u = 0; u < 4; ++u) {
            int n = (wave*4 + u)*16 + n16;      // n>=500 -> weights & bias zero -> stores 0
#pragma unroll
            for (int r = 0; r < 8; ++r) {
                float v = acc[m][u][r] + bf1l[n];
                h1[(m*16 + r + half*8)*520 + n] = (_Float16)(v > 0.f ? v : 0.f);
            }
        }
    __syncthreads();

    if (wave < 2) {                              // fc2: 16 WMMA per mtile
        int m = wave;
        v8f acc2 = {};
        for (int kstep = 0; kstep < 16; ++kstep) {
            const _Float16* ap = h1 + (m*16 + row)*520 + kstep*32 + koff;
            v16h a = cat8(*(const v8h*)ap, *(const v8h*)(ap + 16));
            v16h bb = *(const v16h*)(bf2s + (size_t)kstep*512 + lane*16);
            acc2 = WMMA16(a, bb, acc2);
        }
        if (n16 < 10) {
#pragma unroll
            for (int r = 0; r < 8; ++r)
                out[(size_t)(b0 + m*16 + r + half*8)*10 + n16] = acc2[r] + bf2l[n16];
        }
    }
}

// =========================================================
extern "C" void kernel_launch(void* const* d_in, const int* in_sizes, int n_in,
                              void* d_out, int out_size, void* d_ws, size_t ws_size,
                              hipStream_t stream)
{
    (void)in_sizes; (void)n_in; (void)out_size; (void)ws_size;
    const float* x   = (const float*)d_in[0];
    const float* ps  = (const float*)d_in[1];
    const float* c1w = (const float*)d_in[2];
    const float* c1b = (const float*)d_in[3];
    const float* c2w = (const float*)d_in[4];
    const float* c2b = (const float*)d_in[5];
    const float* f1w = (const float*)d_in[6];
    const float* f1b = (const float*)d_in[7];
    const float* f2w = (const float*)d_in[8];
    const float* f2b = (const float*)d_in[9];

    char* ws = (char*)d_ws;
    _Float16* w1s  = (_Float16*)(ws + O_W1S);
    float*    b1   = (float*)   (ws + O_B1);
    _Float16* b2s  = (_Float16*)(ws + O_B2S);
    float*    b2   = (float*)   (ws + O_B2);
    _Float16* bf1s = (_Float16*)(ws + O_BF1S);
    float*    bf1  = (float*)   (ws + O_BF1);
    _Float16* bf2s = (_Float16*)(ws + O_BF2S);
    float*    bf2  = (float*)   (ws + O_BF2);
    _Float16* y1   = (_Float16*)(ws + O_Y1);
    _Float16* y2   = (_Float16*)(ws + O_Y2);

    prep_kernel<<<(PREP_ELEMS + 255)/256, 256, 0, stream>>>(
        ps, c1w, c1b, c2w, c2b, f1w, f1b, f2w, f2b,
        w1s, b1, b2s, b2, bf1s, bf1, bf2s, bf2);
    conv1_kernel<<<8192, 256, 0, stream>>>(x, w1s, b1, y1);
    conv2_kernel<<<4096, 256, 0, stream>>>(y1, b2s, b2, y2);
    fc_kernel<<<256, 256, 0, stream>>>(y2, bf1s, bf1, bf2s, bf2, (float*)d_out);
}
